// AttentionMeta_58196806861321
// MI455X (gfx1250) — matrix operand order repaired and checked by reading, not compiled, not run on hardware
//
#include <hip/hip_runtime.h>

// ---------------------------------------------------------------------------
// B=1, S=512, D=256 swishmax attention-meta on MI455X (gfx1250, wave32).
//
//   raw[sk] @ Wl == q @ (diag(k[sk,:]) * Wl)   -> per-sk GEMM, stationary q.
//
// Kernel plan (the [Sk,Sq,D] 268MB tensor is never materialized):
//   1) prep GEMMs: k/q/v = X + X@W + b (WMMA f16, f32 accumulate).
//   2) stats kernel: one wg per sk; wave w owns d-columns [16w,16w+16) for
//      ALL sq; B-panel diag(k)*Wl lives in 64 VGPRs; online max / sum|x*e^x|
//      over sq entirely in registers + one shfl_xor(16). Writes m, 1/(t+1).
//   3) output kernel: 16 d-tiles x 8 sk-chunks; 32 waves each own one
//      sq-tile with the full q A-panel in registers; per-sk B panel staged
//      into double-buffered LDS; v[sk,d]*scale accumulates over 64 sk in
//      registers -> atomics reduced 64x (1M global_atomic_add_f32 total).
//   4) final GEMM: out = vsum + vsum@Wvo + bvo.
// ---------------------------------------------------------------------------

typedef __attribute__((ext_vector_type(16))) _Float16 v16h;
typedef __attribute__((ext_vector_type(8)))  _Float16 v8h;
typedef __attribute__((ext_vector_type(8)))  float    v8f;

static constexpr int kS = 512;   // sequence length
static constexpr int kD = 256;   // model dim
static constexpr int kChunks = 8;            // sk chunks in output kernel
static constexpr int kChunk  = kS / kChunks; // 64 sk per chunk
static constexpr int kBpad   = 264;          // LDS row stride in halves (pad)
#define NEG_BIG (-3.0e38f)

static __device__ __forceinline__ int lane_id() { return threadIdx.x & 31; }

static __device__ __forceinline__ v8f zero_v8f() {
  v8f c;
#pragma unroll
  for (int i = 0; i < 8; ++i) c[i] = 0.0f;
  return c;
}

// A-operand 16x32 f16 tile, row-major f16 source:
// lane holds row M = lane&15; half i -> K = k0 + (lane>>4)*16 + i.
// Two 16-byte loads.
static __device__ __forceinline__ v16h load_a_f16(const _Float16* __restrict__ A,
                                                  int row0, int k0, int ld) {
  const int lane = lane_id();
  const int r = lane & 15, h = lane >> 4;
  const _Float16* p = A + (size_t)(row0 + r) * ld + k0 + h * 16;
  v8h lo = *reinterpret_cast<const v8h*>(p);
  v8h hi = *reinterpret_cast<const v8h*>(p + 8);
  return __builtin_shufflevector(lo, hi, 0, 1, 2, 3, 4, 5, 6, 7,
                                 8, 9, 10, 11, 12, 13, 14, 15);
}

// Same A layout, converting from an f32 source.
static __device__ __forceinline__ v16h load_a_f32(const float* __restrict__ A,
                                                  int row0, int k0, int ld) {
  const int lane = lane_id();
  const int r = lane & 15, h = lane >> 4;
  const float* p = A + (size_t)(row0 + r) * ld + k0 + h * 16;
  v16h a;
#pragma unroll
  for (int i = 0; i < 8; ++i) {
    a[i]     = (_Float16)p[i];
    a[i + 8] = (_Float16)p[8 + i];
  }
  return a;
}

// B-operand 32x16 f16 tile from row-major f32 W (column gather):
// lane holds column N = n0 + (lane&15); half i -> K = k0 + (lane>>4)*16 + i.
static __device__ __forceinline__ v16h load_b_f32(const float* __restrict__ W,
                                                  int k0, int n0, int ld) {
  const int lane = lane_id();
  const int h = lane >> 4;
  const int n = n0 + (lane & 15);
  v16h b;
#pragma unroll
  for (int i = 0; i < 16; ++i) {
    const int K = k0 + h * 16 + i;
    b[i] = (_Float16)W[(size_t)K * ld + n];
  }
  return b;
}

// B-operand with per-K row scaling: B = diag(scale) * W  (Wlk = diag(k)*Wl).
static __device__ __forceinline__ v16h load_b_f32_scaled(
    const float* __restrict__ W, const float* __restrict__ scale,
    int k0, int n0, int ld) {
  const int lane = lane_id();
  const int h = lane >> 4;
  const int n = n0 + (lane & 15);
  v16h b;
#pragma unroll
  for (int i = 0; i < 16; ++i) {
    const int K = k0 + h * 16 + i;
    b[i] = (_Float16)(scale[K] * W[(size_t)K * ld + n]);
  }
  return b;
}

// B-operand from LDS panel laid out [16 N][kBpad K] (16 contiguous halves).
static __device__ __forceinline__ v16h load_b_lds(const _Float16* Bs, int ks) {
  const int lane = lane_id();
  const _Float16* p = Bs + (lane & 15) * kBpad + ks * 32 + (lane >> 4) * 16;
  v8h lo = *reinterpret_cast<const v8h*>(p);       // 16B-aligned
  v8h hi = *reinterpret_cast<const v8h*>(p + 8);
  return __builtin_shufflevector(lo, hi, 0, 1, 2, 3, 4, 5, 6, 7,
                                 8, 9, 10, 11, 12, 13, 14, 15);
}

static __device__ __forceinline__ v8f wmma_f16(v16h a, v16h b, v8f c) {
  return __builtin_amdgcn_wmma_f32_16x16x32_f16(
      false, a, false, b, (short)0, c, false, false);
}

// ---------------------------------------------------------------------------
// Y = X + X@W + b   (512x256 @ 256x256). 8 waves/block, one 16x16 tile each.
// ---------------------------------------------------------------------------
__global__ void __launch_bounds__(256) gemm_residual_kernel(
    const float* __restrict__ X, const float* __restrict__ W,
    const float* __restrict__ bias, float* __restrict__ Yf,
    _Float16* __restrict__ Yh) {
  const int wave = threadIdx.x >> 5;
  const int lane = threadIdx.x & 31;
  const int tile = blockIdx.x * 8 + wave;   // 0..511
  const int row0 = (tile >> 4) * 16;
  const int n0   = (tile & 15) * 16;
  const int N    = n0 + (lane & 15);
  const int mofs = (lane >> 4) * 8;

  v8f c = zero_v8f();
#pragma unroll
  for (int ks = 0; ks < 8; ++ks) {
    v16h a = load_a_f32(X, row0, ks * 32, kD);
    v16h b = load_b_f32(W, ks * 32, n0, kD);
    c = wmma_f16(a, b, c);
  }
  const float bN = bias[N];
#pragma unroll
  for (int g = 0; g < 8; ++g) {
    const int M = row0 + mofs + g;
    const float y = c[g] + X[(size_t)M * kD + N] + bN;
    Yf[(size_t)M * kD + N] = y;
    if (Yh) Yh[(size_t)M * kD + N] = (_Float16)y;
  }
}

__global__ void __launch_bounds__(256) zero_kernel(float* __restrict__ p, int n) {
  const int i = blockIdx.x * 256 + threadIdx.x;
  if (i < n) p[i] = 0.0f;
}

// ---------------------------------------------------------------------------
// Stats: one wg per sk; per-(sk,d) online max and sum|l*exp(l-m)| over sq.
// Writes m and inv = 1/(t + NOT_EPSILON) to workspace.
// ---------------------------------------------------------------------------
__global__ void __launch_bounds__(512) attn_stats_kernel(
    const float* __restrict__ kmat, const float* __restrict__ qmat,
    const _Float16* __restrict__ qh, const float* __restrict__ Wl,
    const float* __restrict__ bl, float* __restrict__ m_ws,
    float* __restrict__ inv_ws) {
  const int sk   = blockIdx.x;
  const int wave = threadIdx.x >> 5;
  const int lane = threadIdx.x & 31;
  const int n0   = wave * 16;
  const int N    = n0 + (lane & 15);
  const int mofs = (lane >> 4) * 8;

  const float* krow = kmat + (size_t)sk * kD;
  const float  blN  = bl[N];
  const float  kN   = krow[N];

  // Per-sk B panel diag(k)*Wl [256 x 16] in f16: 64 VGPRs, loaded once.
  v16h Breg[8];
#pragma unroll
  for (int ks = 0; ks < 8; ++ks)
    Breg[ks] = load_b_f32_scaled(Wl, krow, ks * 32, n0, kD);

  float m = NEG_BIG;   // running max of logits over this (sk,d) column
  float t = 0.0f;      // running sum of |l|*exp(l - m)
  for (int st = 0; st < kS / 16; ++st) {
    if (st + 1 < kS / 16)
      __builtin_prefetch(qh + (size_t)(st + 1) * 16 * kD, 0, 0);
    v8f c = zero_v8f();
#pragma unroll
    for (int ks = 0; ks < 8; ++ks)
      c = wmma_f16(load_a_f16(qh, st * 16, ks * 32, kD), Breg[ks], c);
#pragma unroll
    for (int g = 0; g < 8; ++g) {
      const int M = st * 16 + mofs + g;
      const float l = c[g] + kN * qmat[(size_t)M * kD + N] + blN;
      const float mn = fmaxf(m, l);
      t = t * __expf(m - mn) + fabsf(l) * __expf(l - mn);
      m = mn;
    }
  }
  // Lanes L and L+16 hold disjoint sq rows of the same column N: merge.
  const float m2 = __shfl_xor(m, 16);
  const float t2 = __shfl_xor(t, 16);
  const float mm = fmaxf(m, m2);
  t = t * __expf(m - mm) + t2 * __expf(m2 - mm);
  m = mm;

  if (lane < 16) {
    m_ws[(size_t)sk * kD + N]   = m;
    inv_ws[(size_t)sk * kD + N] = 1.0f / (t + 1.0f);   // NOT_EPSILON = 1.0
  }
}

// ---------------------------------------------------------------------------
// Output: grid = 16 d-tiles x kChunks sk-chunks; 1024 threads = 32 waves,
// wave w owns sq-tile w with its full q A-panel (K=256) in 64 VGPRs.
// Per-sk B panel staged in double-buffered LDS; v*scale accumulates in
// registers over kChunk sk values before one atomic per element.
// ---------------------------------------------------------------------------
__global__ void __launch_bounds__(1024) attn_out_kernel(
    const float* __restrict__ kmat, const float* __restrict__ qmat,
    const _Float16* __restrict__ qh, const float* __restrict__ vmat,
    const float* __restrict__ Wl, const float* __restrict__ bl,
    const float* __restrict__ m_ws, const float* __restrict__ inv_ws,
    float* __restrict__ vsum) {
  __shared__ _Float16 Bs[2][16 * kBpad];   // 2 x 8.25KB panels

  const int dt   = blockIdx.x & 15;        // d tile
  const int ck   = blockIdx.x >> 4;        // sk chunk
  const int n0   = dt * 16;
  const int wave = threadIdx.x >> 5;       // 0..31 -> sq tile
  const int lane = threadIdx.x & 31;
  const int row0 = wave * 16;
  const int N    = n0 + (lane & 15);
  const int mofs = (lane >> 4) * 8;
  const int tid  = threadIdx.x;
  const int sn   = tid & 15;               // staging: column within d tile
  const int kr   = tid >> 4;               // staging: K row 0..63 (+64*j)

  // Full A panel for this wave's sq tile (constant across sk): 64 VGPRs.
  v16h Apanel[8];
#pragma unroll
  for (int ks = 0; ks < 8; ++ks) Apanel[ks] = load_a_f16(qh, row0, ks * 32, kD);

  // Per-(sq,d) terms constant across sk.
  const float blN = bl[N];
  float qraw[8];
#pragma unroll
  for (int g = 0; g < 8; ++g)
    qraw[g] = qmat[(size_t)(row0 + mofs + g) * kD + N];

  v8f acc = zero_v8f();
  const int sk0 = ck * kChunk;

  // Stage first B panel (diag(k[sk0])*Wl columns [n0,n0+16)) into buffer 0.
#pragma unroll
  for (int j = 0; j < 4; ++j) {
    const int K = kr + 64 * j;
    Bs[0][sn * kBpad + K] =
        (_Float16)(kmat[(size_t)sk0 * kD + K] * Wl[(size_t)K * kD + n0 + sn]);
  }
  __syncthreads();

  for (int i = 0; i < kChunk; ++i) {
    const int sk  = sk0 + i;
    const int cur = i & 1;

    // Issue next panel's global loads early (overlap with WMMA latency).
    float nb[4];
    if (i + 1 < kChunk) {
#pragma unroll
      for (int j = 0; j < 4; ++j) {
        const int K = kr + 64 * j;
        nb[j] = kmat[(size_t)(sk + 1) * kD + K] * Wl[(size_t)K * kD + n0 + sn];
      }
    }

    // Per-sk scalars for this lane's column N (coalesced).
    const float kN  = kmat[(size_t)sk * kD + N];
    const float vN  = vmat[(size_t)sk * kD + N];
    const float m   = m_ws[(size_t)sk * kD + N];
    const float inv = inv_ws[(size_t)sk * kD + N];

    v8f c = zero_v8f();
#pragma unroll
    for (int ks = 0; ks < 8; ++ks)
      c = wmma_f16(Apanel[ks], load_b_lds(Bs[cur], ks), c);

#pragma unroll
    for (int g = 0; g < 8; ++g) {
      const float l  = c[g] + kN * qraw[g] + blN;
      const float sc = l * __expf(l - m) * inv;   // swishmax scale
      acc[g] += vN * sc;
    }

    if (i + 1 < kChunk) {
#pragma unroll
      for (int j = 0; j < 4; ++j)
        Bs[1 - cur][sn * kBpad + (kr + 64 * j)] = (_Float16)nb[j];
    }
    __syncthreads();
  }

#pragma unroll
  for (int g = 0; g < 8; ++g)
    unsafeAtomicAdd(&vsum[(size_t)(row0 + mofs + g) * kD + N], acc[g]);
}

// ---------------------------------------------------------------------------
extern "C" void kernel_launch(void* const* d_in, const int* in_sizes, int n_in,
                              void* d_out, int out_size, void* d_ws, size_t ws_size,
                              hipStream_t stream) {
  const float* query = (const float*)d_in[0];
  const float* key   = (const float*)d_in[1];
  const float* value = (const float*)d_in[2];
  const float* Wk    = (const float*)d_in[3];
  const float* bk    = (const float*)d_in[4];
  const float* Wq    = (const float*)d_in[5];
  const float* bq    = (const float*)d_in[6];
  const float* Wva   = (const float*)d_in[7];
  const float* bva   = (const float*)d_in[8];
  const float* Wl    = (const float*)d_in[9];
  const float* bl    = (const float*)d_in[10];
  const float* Wvo   = (const float*)d_in[11];
  const float* bvo   = (const float*)d_in[12];

  // Workspace layout (~3.25 MB total).
  char* ws = (char*)d_ws;
  float*    kbuf   = (float*)ws;  ws += (size_t)kS * kD * sizeof(float);
  float*    qbuf   = (float*)ws;  ws += (size_t)kS * kD * sizeof(float);
  float*    vbuf   = (float*)ws;  ws += (size_t)kS * kD * sizeof(float);
  float*    vsum   = (float*)ws;  ws += (size_t)kS * kD * sizeof(float);
  float*    m_ws   = (float*)ws;  ws += (size_t)kS * kD * sizeof(float);
  float*    inv_ws = (float*)ws;  ws += (size_t)kS * kD * sizeof(float);
  _Float16* qh     = (_Float16*)ws;  // kS*kD halves

  const int n = kS * kD;
  zero_kernel<<<(n + 255) / 256, 256, 0, stream>>>(vsum, n);

  // k/q/v residual GEMMs (q also emitted as f16 A-operand copy).
  gemm_residual_kernel<<<64, 256, 0, stream>>>(key,   Wk,  bk,  kbuf, nullptr);
  gemm_residual_kernel<<<64, 256, 0, stream>>>(query, Wq,  bq,  qbuf, qh);
  gemm_residual_kernel<<<64, 256, 0, stream>>>(value, Wva, bva, vbuf, nullptr);

  // Swishmax statistics per (sk,d), then register-accumulated output.
  attn_stats_kernel<<<kS, 512, 0, stream>>>(kbuf, qbuf, qh, Wl, bl, m_ws, inv_ws);
  attn_out_kernel<<<16 * kChunks, 1024, 0, stream>>>(
      kbuf, qbuf, qh, vbuf, Wl, bl, m_ws, inv_ws, vsum);

  // out = vsum + vsum@Wvo + bvo
  gemm_residual_kernel<<<64, 256, 0, stream>>>(vsum, Wvo, bvo, (float*)d_out, nullptr);
}
